// SharedSelfAttentionV0_54554674593973
// MI455X (gfx1250) — matrix operand order repaired and checked by reading, not compiled, not run on hardware
//
#include <hip/hip_runtime.h>

// ---------------------------------------------------------------------------
// Hyperbolic GNN attention layer for MI455X (gfx1250, wave32, WMMA + TDM).
// N=20000 nodes, D=128, DEG=32 (edges for node i are [i*32, i*32+32)).
// Round 3: Tensor Data Mover staging of the L tile into LDS (guarded),
// pre-packed f16 B fragments for WMMA, b128 per-lane gathers.
// ---------------------------------------------------------------------------

typedef __attribute__((ext_vector_type(16))) _Float16 v16h;
typedef __attribute__((ext_vector_type(4)))  _Float16 v4h;
typedef __attribute__((ext_vector_type(8)))  float    v8f;
typedef __attribute__((ext_vector_type(4)))  unsigned int v4u;
typedef __attribute__((ext_vector_type(4)))  int      v4i;
typedef __attribute__((ext_vector_type(8)))  int      v8i;

#define NN    20000
#define DEG   32
#define D     128
#define MINN  1e-15f
#define EPSF  1e-7f

__device__ __forceinline__ float wsum(float v) {
#pragma unroll
  for (int o = 16; o > 0; o >>= 1) v += __shfl_xor(v, o, 32);
  return v;
}

__device__ __forceinline__ float acoshf_(float x) {
  // caller guarantees x >= 1+eps
  return logf(x + sqrtf(x * x - 1.0f));
}

// ---------------------------------------------------------------------------
// Precompute lb[k] = logmap0(proj(expmap0([0, b[1:]]))) for the 3 biases.
// One wave per bias; lane handles components j = q*32 + lane.
// ---------------------------------------------------------------------------
__global__ __launch_bounds__(32) void bias_prep_kernel(
    const float* __restrict__ b_att, const float* __restrict__ b_data,
    const float* __restrict__ b_out, float* __restrict__ lb) {
  const int bid  = blockIdx.x;
  const float* b = (bid == 0) ? b_att : (bid == 1) ? b_data : b_out;
  const int lane = threadIdx.x;

  float bv[4], part = 0.f;
#pragma unroll
  for (int q = 0; q < 4; ++q) {
    int j = q * 32 + lane;
    bv[q] = (j == 0) ? 0.f : b[j];
    part += bv[q] * bv[q];
  }
  float xn = fmaxf(sqrtf(wsum(part)), MINN);      // theta
  float sh = sinhf(xn);
  float rt[4];
  part = 0.f;
#pragma unroll
  for (int q = 0; q < 4; ++q) {                   // expmap0 tail
    rt[q] = sh * bv[q] / xn;
    part += rt[q] * rt[q];
  }
  float ss = wsum(part);
  float h0 = sqrtf(1.f + ss);                     // proj component 0
  // logmap0
  float yn  = fmaxf(sqrtf(ss), MINN);
  float fac = acoshf_(fmaxf(h0, 1.f + EPSF)) / yn;
#pragma unroll
  for (int q = 0; q < 4; ++q) {
    int j = q * 32 + lane;
    lb[bid * D + j] = (j == 0) ? 0.f : fac * rt[q];
  }
}

// ---------------------------------------------------------------------------
// One-shot pack of the three 128x128 f32 weight matrices into per-lane WMMA
// B fragments (f16). Fragment index within a matrix:
//   frag = (ntile*4 + kt)*32 + lane, 16 halfs:
//   half j -> K = kt*32 + j + (lane>=16 ? 16 : 0), N = ntile*16 + (lane&15),
//   value = W[N][K]   (B = W^T).
// 3 mats * 8 ntiles * 4 kt * 32 lanes = 3072 fragments; one per thread.
// ---------------------------------------------------------------------------
__global__ __launch_bounds__(256) void pack_w_kernel(
    const float* __restrict__ W_att, const float* __restrict__ W_data,
    const float* __restrict__ W_out, _Float16* __restrict__ Bp) {
  const int frag  = blockIdx.x * 256 + threadIdx.x;   // 0..3071
  const int lane  = frag & 31;
  const int kt    = (frag >> 5) & 3;
  const int ntile = (frag >> 7) & 7;
  const int mat   = frag >> 10;
  const float* W  = (mat == 0) ? W_att : (mat == 1) ? W_data : W_out;

  const int n  = ntile * 16 + (lane & 15);
  const int hi = lane >> 4;
  v16h b;
#pragma unroll
  for (int j = 0; j < 16; ++j) {
    int k = kt * 32 + j + hi * 16;
    b[j] = (_Float16)W[(size_t)n * D + k];
  }
  ((v16h*)Bp)[frag] = b;
}

// ---------------------------------------------------------------------------
// logmap0 of the input hyperboloid features -> f16 (WMMA A operand).
// Wave per node (8 nodes / 256-thread block); b128 per lane (j = lane*4+q).
// ---------------------------------------------------------------------------
__global__ __launch_bounds__(256) void logmap_feat_kernel(
    const float* __restrict__ x, _Float16* __restrict__ L) {
  const int lane = threadIdx.x & 31;
  const int node = blockIdx.x * 8 + (threadIdx.x >> 5);

  float4 x4 = ((const float4*)x)[(size_t)node * 32 + lane];
  float xv[4] = {x4.x, x4.y, x4.z, x4.w};
  float part = 0.f;
#pragma unroll
  for (int q = 0; q < 4; ++q) {
    bool is0 = (lane == 0) && (q == 0);
    part += is0 ? 0.f : xv[q] * xv[q];
  }
  float x0  = __shfl(xv[0], 0, 32);               // component 0 = lane0.q0
  float yn  = fmaxf(sqrtf(wsum(part)), MINN);
  float fac = acoshf_(fmaxf(x0, 1.f + EPSF)) / yn;
  v4h o;
#pragma unroll
  for (int q = 0; q < 4; ++q) {
    bool is0 = (lane == 0) && (q == 0);
    o[q] = is0 ? (_Float16)0.f : (_Float16)(fac * xv[q]);
  }
  ((v4h*)L)[(size_t)node * 32 + lane] = o;
}

// ---------------------------------------------------------------------------
// hyp_linear: Y = L @ W^T via V_WMMA_F32_16X16X32_F16 with pre-packed B
// fragments, then per-node expmap0 -> mobius_add(bias) -> proj.
// Block = 256 threads = 8 waves, 16 nodes per block. Wave w owns N-tile
// [16w,16w+16), K-loop of 4 WMMAs, then postprocesses nodes 2w, 2w+1.
// The 16x128 f16 L tile (4 KB contiguous) is DMA'd into LDS by the Tensor
// Data Mover when the builtin is available (TENSORcnt-tracked), else staged
// with one float4 per thread.
// mode 0: store hyperboloid point.  mode 1: store to_poincare (slot0 = 0).
// mode 2: store hyperboloid point to final output.
// ---------------------------------------------------------------------------
__global__ __launch_bounds__(256) void hyp_linear_kernel(
    const _Float16* __restrict__ Lsrc, const v16h* __restrict__ Bp,
    const float* __restrict__ lb, float* __restrict__ out, int mode) {
  __shared__ __align__(16) _Float16 sL[16 * D];
  __shared__ __align__(16) float sY[16][D];

  const int tid   = threadIdx.x;
  const int wave  = tid >> 5;
  const int lane  = tid & 31;
  const int node0 = blockIdx.x * 16;

  const _Float16* gsrc = Lsrc + (size_t)node0 * D;

#if defined(__has_builtin) && __has_builtin(__builtin_amdgcn_tensor_load_to_lds)
  // TDM: one 4 KB contiguous 2D-degenerate tile, issued by wave 0 only
  // (TDM ignores EXEC; uniform branch keeps other waves from issuing).
  if (wave == 0) {
    unsigned long long ga = (unsigned long long)(size_t)(const void*)gsrc;
    unsigned lds_off = (unsigned)(size_t)(const void*)&sL[0];  // addr[31:0] = LDS offset
    v4u g0;
    g0[0] = 1u;                                          // count=1, user D#
    g0[1] = lds_off;                                     // lds_addr
    g0[2] = (unsigned)(ga & 0xffffffffu);                // global_addr[31:0]
    g0[3] = (unsigned)((ga >> 32) & 0x01ffffffu) | 0x80000000u;  // addr[56:32] | type=2
    v8i g1;
    g1[0] = 0x00010000;                                  // data_size=1 (2 B), no flags
    g1[1] = (int)(2048u << 16);                          // tensor_dim0[15:0] = 2048
    g1[2] = (int)(1u << 16);                             // tensor_dim0 hi=0 | tensor_dim1=1
    g1[3] = (int)(2048u << 16);                          // tile_dim0 = 2048
    g1[4] = 1;                                           // tile_dim1 = 1, tile_dim2 = 0
    g1[5] = 2048;                                        // tensor_dim0_stride[31:0]
    g1[6] = 0;
    g1[7] = 0;
    v4i gz = {0, 0, 0, 0};
#if __clang_major__ >= 23
    v8i gz8 = {0, 0, 0, 0, 0, 0, 0, 0};
    __builtin_amdgcn_tensor_load_to_lds(g0, g1, gz, gz, gz8, 0);
#else
    __builtin_amdgcn_tensor_load_to_lds(g0, g1, gz, gz, 0);
#endif
#if __has_builtin(__builtin_amdgcn_s_wait_tensorcnt)
    __builtin_amdgcn_s_wait_tensorcnt(0);
#endif
  }
  __syncthreads();
#else
  ((float4*)sL)[tid] = ((const float4*)gsrc)[tid];
  __syncthreads();
#endif

  // --- WMMA: D = A(16x32) x B(32x16) + C, K-loop of 4 ---
  const int mrow = lane & 15;
  const int hi   = lane >> 4;                 // lane half selects K sub-block
  const int n    = wave * 16 + (lane & 15);   // global output column
  v8f acc = {};
#pragma unroll
  for (int kt = 0; kt < 4; ++kt) {
    const int kk = kt * 32;
    v16h a;
    // A fragment: contiguous K order matching packed B: K=j,
    // upper lanes offset K by 16.
#pragma unroll
    for (int j = 0; j < 16; ++j) {
      int ka = kk + j + hi * 16;
      a[j] = sL[mrow * D + ka];
    }
    // B fragment: pre-packed f16, one 32-byte load.
    v16h b = Bp[(size_t)(wave * 4 + kt) * 32 + lane];
    acc = __builtin_amdgcn_wmma_f32_16x16x32_f16(
        /*neg_a=*/false, a, /*neg_b=*/false, b,
        /*c_mod=*/(short)0, acc, /*reuse_a=*/false, /*reuse_b=*/false);
  }
  // Scatter C/D fragment: VGPR r -> row M = r + 8*hi, col n.
#pragma unroll
  for (int r = 0; r < 8; ++r) sY[r + 8 * hi][n] = acc[r];
  __syncthreads();

  // --- Per-node hyperbolic postprocess: wave handles nodes 2w, 2w+1 ---
  // Component mapping j = lane*4 + q; component 0 lives at lane 0, q 0.
#pragma unroll 1
  for (int s = 0; s < 2; ++s) {
    const int m = wave * 2 + s;
    float4 y4 = ((const float4*)sY[m])[lane];
    float y[4] = {y4.x, y4.y, y4.z, y4.w};
    if (lane == 0) y[0] = 0.f;                // expmap0 ignores component 0
    float part = 0.f;
#pragma unroll
    for (int q = 0; q < 4; ++q) part += y[q] * y[q];
    float xn = fmaxf(sqrtf(wsum(part)), MINN);
    float sh = sinhf(xn);
    float res[4];
    part = 0.f;
#pragma unroll
    for (int q = 0; q < 4; ++q) {
      res[q] = sh * y[q] / xn;
      part += res[q] * res[q];
    }
    float ss   = wsum(part);
    float res0 = sqrtf(1.f + ss);             // proj

    // mobius_add(res, hyp_bias) with lb = logmap0(hyp_bias) precomputed.
    float yn = fmaxf(sqrtf(ss), MINN);
    float4 lb4 = ((const float4*)lb)[lane];   // lb[0] == 0 by construction
    float lbv[4] = {lb4.x, lb4.y, lb4.z, lb4.w};
    float yhat[4];
    part = 0.f;
#pragma unroll
    for (int q = 0; q < 4; ++q) {
      yhat[q] = res[q] / yn;
      part   += yhat[q] * lbv[q];
    }
    float alpha = wsum(part);
    float w_[4];
    float omr = 1.f - res0;
    part = 0.f;
#pragma unroll
    for (int q = 0; q < 4; ++q) {
      bool is0 = (lane == 0) && (q == 0);
      float vv = is0 ? -yn : omr * yhat[q];
      w_[q]    = lbv[q] - alpha * vv;
      part    += res[q] * w_[q];              // res[0]==0 -> tail dot only
    }
    float ux = wsum(part);
    float t0 = ux / fmaxf(res0, EPSF);        // proj_tan component 0
    float t[4];
    part = 0.f;
#pragma unroll
    for (int q = 0; q < 4; ++q) {
      bool is0 = (lane == 0) && (q == 0);
      t[q] = is0 ? t0 : w_[q];
      part += t[q] * t[q] * (is0 ? -1.f : 1.f);       // mink_dot(t,t)
    }
    float md  = wsum(part);
    float th2 = fmaxf(sqrtf(fmaxf(md, 0.f)), MINN);
    float ch  = coshf(th2), shk = sinhf(th2) / th2;
    float z[4];
    part = 0.f;
#pragma unroll
    for (int q = 0; q < 4; ++q) {
      bool is0 = (lane == 0) && (q == 0);
      float xb = is0 ? res0 : res[q];
      z[q]     = ch * xb + shk * t[q];        // expmap
      part    += is0 ? 0.f : z[q] * z[q];
    }
    float z0 = sqrtf(1.f + wsum(part));       // final proj

    float4 fo;
    float fv[4];
#pragma unroll
    for (int q = 0; q < 4; ++q) {
      bool is0 = (lane == 0) && (q == 0);
      fv[q] = is0 ? z0 : z[q];
      if (mode == 1) fv[q] = is0 ? 0.f : z[q] / (z0 + 1.f);   // to_poincare
    }
    fo.x = fv[0]; fo.y = fv[1]; fo.z = fv[2]; fo.w = fv[3];
    ((float4*)(out + (size_t)(node0 + m) * D))[lane] = fo;
  }
}

// ---------------------------------------------------------------------------
// Edge phase: wave per node. Pass 1: attention coefs + denom; Pass 2:
// Poincare-weighted aggregation; then mobius_scalar_mul(0.5)/from_poincare/
// logmap0 -> f16 input for the final GEMM. All row accesses are one b128
// per lane (j = lane*4 + q).
// ---------------------------------------------------------------------------
__global__ __launch_bounds__(256) void edge_kernel(
    const float* __restrict__ att, const float* __restrict__ vbuf,
    const float* __restrict__ eval, const int* __restrict__ etgt,
    _Float16* __restrict__ Lmid) {
  const int lane  = threadIdx.x & 31;
  const int node  = blockIdx.x * 8 + (threadIdx.x >> 5);
  const int ebase = node * DEG;
  const float4* attv = (const float4*)att;
  const float4* vbv  = (const float4*)vbuf;

  // Source att row, pre-signed for Minkowski dot (component 0 gets -1).
  float4 a4 = attv[(size_t)node * 32 + lane];
  float asr[4] = {a4.x, a4.y, a4.z, a4.w};
  if (lane == 0) asr[0] = -asr[0];

  float csave = 0.f, c2 = 0.f;
  int tsave = 0;
#pragma unroll 1
  for (int e = 0; e < DEG; ++e) {
    int tgt = etgt[ebase + e];
    if (e + 1 < DEG) __builtin_prefetch(att + (size_t)etgt[ebase + e + 1] * D, 0, 1);
    float4 t4 = attv[(size_t)tgt * 32 + lane];
    float p = t4.x * asr[0] + t4.y * asr[1] + t4.z * asr[2] + t4.w * asr[3];
    float mdot = wsum(p);                     // mink_dot(att[tgt], att[src])
    float thv  = fmaxf(-mdot, 1.f + EPSF);
    float dd   = acoshf_(thv);
    float sq   = fminf(dd * dd, 50.f);
    float coef = -eval[ebase + e] * sq;
    c2 += coef * coef;
    if (lane == e) { csave = coef; tsave = tgt; }
  }
  float denom = fmaxf(sqrtf(c2), 1e-12f);

  float num[4] = {0.f, 0.f, 0.f, 0.f};
  float den = 0.f;
#pragma unroll 1
  for (int e = 0; e < DEG; ++e) {
    float coef = __shfl(csave, e, 32);
    int tgt    = __shfl(tsave, e, 32);
    float a    = coef / denom;
    float4 v4 = vbv[(size_t)tgt * 32 + lane];  // slot 0 stored as 0
    float vv[4] = {v4.x, v4.y, v4.z, v4.w};
    float p = vv[0]*vv[0] + vv[1]*vv[1] + vv[2]*vv[2] + vv[3]*vv[3];
    float ssv = wsum(p);
    float g2  = 1.f / fmaxf(1.f - ssv, MINN); // gamma^2
#pragma unroll
    for (int q = 0; q < 4; ++q) num[q] += a * g2 * vv[q];
    den += a * (g2 - 0.5f);
  }
  if (fabsf(den) < MINN) den = MINN;

  // mid = mobius_scalar_mul(0.5, num/den)
  float u[4], part = 0.f;
#pragma unroll
  for (int q = 0; q < 4; ++q) { u[q] = num[q] / den; part += u[q] * u[q]; }
  float un  = fmaxf(sqrtf(wsum(part)), MINN);
  float unc = fminf(un, 1.f - EPSF);
  float at  = 0.5f * logf((1.f + unc) / (1.f - unc));  // artanh
  float fac = tanhf(0.5f * at) / un;
  float mid[4];
  part = 0.f;
#pragma unroll
  for (int q = 0; q < 4; ++q) { mid[q] = fac * u[q]; part += mid[q] * mid[q]; }

  // from_poincare
  float sq  = wsum(part);
  float inv = 1.f / fmaxf(1.f - sq, MINN);
  float h0  = (1.f + sq) * inv;
  float h[4];
  part = 0.f;
#pragma unroll
  for (int q = 0; q < 4; ++q) {
    bool is0 = (lane == 0) && (q == 0);
    h[q] = is0 ? h0 : 2.f * mid[q] * inv;
    part += is0 ? 0.f : h[q] * h[q];
  }
  // logmap0 -> f16 for final GEMM
  float yn  = fmaxf(sqrtf(wsum(part)), MINN);
  float lfc = acoshf_(fmaxf(h0, 1.f + EPSF)) / yn;
  v4h o;
#pragma unroll
  for (int q = 0; q < 4; ++q) {
    bool is0 = (lane == 0) && (q == 0);
    o[q] = is0 ? (_Float16)0.f : (_Float16)(lfc * h[q]);
  }
  ((v4h*)Lmid)[(size_t)node * 32 + lane] = o;
}

// ---------------------------------------------------------------------------
extern "C" void kernel_launch(void* const* d_in, const int* in_sizes, int n_in,
                              void* d_out, int out_size, void* d_ws, size_t ws_size,
                              hipStream_t stream) {
  const float* feats  = (const float*)d_in[0];
  const float* eval   = (const float*)d_in[1];
  const float* W_att  = (const float*)d_in[2];
  const float* b_att  = (const float*)d_in[3];
  const float* W_data = (const float*)d_in[4];
  const float* b_data = (const float*)d_in[5];
  const float* W_out  = (const float*)d_in[6];
  const float* b_out  = (const float*)d_in[7];
  const int*   etgt   = (const int*)d_in[9];   // edge_src is implicit (repeat)
  float* out = (float*)d_out;

  char* ws = (char*)d_ws;
  float*    att   = (float*)(ws);                    // N*D f32 = 10.24 MB
  float*    vbuf  = (float*)(ws + 10240000);         // N*D f32 = 10.24 MB
  _Float16* Lfeat = (_Float16*)(ws + 20480000);      // N*D f16 =  5.12 MB
  _Float16* Lmid  = (_Float16*)(ws + 25600000);      // N*D f16 =  5.12 MB
  float*    lb    = (float*)(ws + 30720000);         // 3*D f32 (2 KB slot)
  _Float16* Bp    = (_Float16*)(ws + 30722048);      // 3*1024 frags * 32 B

  bias_prep_kernel<<<3, 32, 0, stream>>>(b_att, b_data, b_out, lb);
  pack_w_kernel<<<12, 256, 0, stream>>>(W_att, W_data, W_out, Bp);
  logmap_feat_kernel<<<NN / 8, 256, 0, stream>>>(feats, Lfeat);

  const v16h* Bpv = (const v16h*)Bp;
  hyp_linear_kernel<<<NN / 16, 256, 0, stream>>>(Lfeat, Bpv,        lb,         att, 0);
  hyp_linear_kernel<<<NN / 16, 256, 0, stream>>>(Lfeat, Bpv + 1024, lb + D,     vbuf, 1);
  edge_kernel<<<NN / 8, 256, 0, stream>>>(att, vbuf, eval, etgt, Lmid);
  hyp_linear_kernel<<<NN / 16, 256, 0, stream>>>(Lmid,  Bpv + 2048, lb + 2 * D, out, 2);
}
